// NeuralODE_71433896067242
// MI455X (gfx1250) — matrix operand order repaired and checked by reading, not compiled, not run on hardware
//
#include <hip/hip_runtime.h>
#include <hip/hip_bf16.h>

// ---------------------------------------------------------------------------
// Tsit5 Neural-ODE integrator for MI455X (gfx1250, wave32, WMMA).
//   - weights (f16, transposed, padded) resident in LDS (~201 KB)
//   - y and k1..k5 resident in VGPRs in WMMA C-layout
//   - all GEMMs via v_wmma_f32_16x16x32_f16, A/B fragments from LDS
//   - N-tiles blocked in groups of 4 so each A fragment feeds 4 WMMAs
//   - staging buffers at LDS offset 0 so all staging ds ops use the 16-bit
//     immediate offset field (no per-store address VALU adds)
//   - 2 waves/block, 128 blocks: one block per WGP on a 128-WGP-class chip,
//     maximizing per-wave LDS bandwidth (the secondary bottleneck)
//   - each wave integrates 16 batch rows independently (no hot-loop barriers)
// ---------------------------------------------------------------------------

typedef __attribute__((ext_vector_type(8)))  _Float16 v8h;
typedef __attribute__((ext_vector_type(16))) _Float16 v16h;
typedef __attribute__((ext_vector_type(8)))  float    v8f;

#define DEV __device__ __forceinline__

// Problem dims
constexpr int NB = 4096, NT = 64, ND = 64, NW = 256;
constexpr int SUBSTEPS = 2;

// Tsit5 tableau
constexpr float A21 = 0.161f;
constexpr float A31 = -0.008480655492356989f, A32 = 0.335480655492357f;
constexpr float A41 = 2.8971530571054935f, A42 = -6.359448489975075f, A43 = 4.3622954328695815f;
constexpr float A51 = 5.325864828439257f, A52 = -11.748883564062828f, A53 = 7.4955393428898365f, A54 = -0.09249506636175525f;
constexpr float A61 = 5.86145544294642f, A62 = -12.92096931784711f, A63 = 8.159367898576159f, A64 = -0.071584973281401f, A65 = -0.028269050394068383f;
constexpr float CB1 = 0.09646076681806523f, CB2 = 0.01f, CB3 = 0.4798896504144996f,
                CB4 = 1.379008574103742f,  CB5 = -3.290069515436081f, CB6 = 2.324710524099774f;

// LDS strides in halves (pad +8 to de-align banks)
constexpr int SW0 = 72;   // W0^T : [256][72]   (K = 64)
constexpr int SW1 = 264;  // W1^T : [256][264]  (K = 256)
constexpr int SW2 = 264;  // W2^T : [64][264]   (K = 256)
constexpr int SX  = 264;  // activation staging [16][264]

constexpr int WAVES = 2;
constexpr int BLOCK = WAVES * 32;
constexpr int ROWS_PER_BLOCK = WAVES * 16;      // 32
constexpr int GRID = NB / ROWS_PER_BLOCK;       // 128

// LDS layout (bytes, every section 16B aligned).
// Staging first (low offsets -> DS 16-bit immediates), big W1 last.
constexpr size_t SZ_BUF  = size_t(16) * SX * 2;            // 8448 per buffer
constexpr size_t OFF_BUF = 0;
constexpr size_t SZ_BUFS = size_t(WAVES) * 2 * SZ_BUF;     // 33792
constexpr size_t OFF_B0  = OFF_BUF + SZ_BUFS;
constexpr size_t OFF_B1  = OFF_B0 + 256 * 4;
constexpr size_t OFF_B2  = OFF_B1 + 256 * 4;
constexpr size_t OFF_W0  = OFF_B2 + 64 * 4;
constexpr size_t SZ_W0   = size_t(256) * SW0 * 2;          // 36864
constexpr size_t OFF_W2  = OFF_W0 + SZ_W0;
constexpr size_t SZ_W2   = size_t(64) * SW2 * 2;           // 33792
constexpr size_t OFF_W1  = OFF_W2 + SZ_W2;
constexpr size_t SZ_W1   = size_t(256) * SW1 * 2;          // 135168
constexpr size_t LDS_TOTAL = OFF_W1 + SZ_W1;               // 241920 B < 320 KB

DEV v16h cat8(v8h a, v8h b) {
  return __builtin_shufflevector(a, b, 0,1,2,3,4,5,6,7,8,9,10,11,12,13,14,15);
}

// A fragment (16x32 f16): lane<16 holds row=lane, K {k0..k0+7, k0+16..k0+23};
// lane>=16 holds row=lane-16, K {k0+8..k0+15, k0+24..k0+31}. Row-major LDS.
template<int S>
DEV v16h loadA(const _Float16* x, int lane, int kt) {
  const int row  = lane & 15;
  const int koff = kt * 32 + ((lane & 16) ? 8 : 0);
  const v8h* p = (const v8h*)(x + row * S + koff);
  return cat8(p[0], p[2]);   // +0 and +16 halves
}

// B fragment (32x16 f16) from W^T [N][K] row-major:
// lane<16: N=n0+lane, K {k0..k0+7, k0+16..k0+23}; lane>=16: N=n0+lane-16, K {k0+8..k0+15, k0+24..k0+31}
template<int S>
DEV v16h loadB(const _Float16* wt, int lane, int n0, int kt) {
  const int n    = n0 + (lane & 15);
  const int koff = kt * 32 + ((lane & 16) ? 8 : 0);
  const v8h* p = (const v8h*)(wt + n * S + koff);
  return cat8(p[0], p[2]);   // +0 and +16 halves
}

DEV v8f wmma_f16(v16h a, v16h b, v8f c) {
  return __builtin_amdgcn_wmma_f32_16x16x32_f16(false, a, false, b, (short)0, c, false, false);
}

// gfx1250 hardware tanh (TRANS op). Prefer the builtin so the compiler
// schedules the TRANS hazard window; else asm + 1 V_NOP.
DEV float tanh_hw(float x) {
#if __has_builtin(__builtin_amdgcn_tanhf)
  return __builtin_amdgcn_tanhf(x);
#elif __has_builtin(__builtin_amdgcn_tanh_f32)
  return __builtin_amdgcn_tanh_f32(x);
#else
  float r;
  asm("v_tanh_f32 %0, %1\n\tv_nop" : "=v"(r) : "v"(x));
  return r;
#endif
}

// Hidden layer: out[16, NGROUPS*64] = tanh(x @ W + b), staged to LDS f16.
// N-tiles processed in groups of 4: one A fragment feeds 4 WMMAs.
template<int NGROUPS, int KTILES, int SWGT>
DEV void layer_hidden(const _Float16* xin, const _Float16* wt, const float* bias,
                      _Float16* xout, int lane) {
  const int col = lane & 15;
  const int rb  = (lane & 16) ? 8 : 0;
  for (int g = 0; g < NGROUPS; ++g) {
    v8f c[4] = {};
#pragma unroll
    for (int kt = 0; kt < KTILES; ++kt) {
      const v16h a = loadA<SX>(xin, lane, kt);
#pragma unroll
      for (int j = 0; j < 4; ++j)
        c[j] = wmma_f16(a, loadB<SWGT>(wt, lane, (g * 4 + j) * 16, kt), c[j]);
    }
#pragma unroll
    for (int j = 0; j < 4; ++j) {
      const float bv = bias[(g * 4 + j) * 16 + col];
#pragma unroll
      for (int r = 0; r < 8; ++r)
        xout[(rb + r) * SX + (g * 4 + j) * 16 + col] = (_Float16)tanh_hw(c[j][r] + bv);
    }
  }
}

// Output layer: k (C-layout VGPRs) = x @ W2 + b2, optionally folded as
// k += alpha * (x@W2+b2) for the k6 stage. Single group of 4 N-tiles.
template<int SWGT>
DEV void layer_out(const _Float16* xin, const _Float16* wt, const float* bias,
                   v8f k[4], float alpha, bool accum, int lane) {
  const int col = lane & 15;
  v8f c[4] = {};
#pragma unroll
  for (int kt = 0; kt < 8; ++kt) {
    const v16h a = loadA<SX>(xin, lane, kt);
#pragma unroll
    for (int j = 0; j < 4; ++j)
      c[j] = wmma_f16(a, loadB<SWGT>(wt, lane, j * 16, kt), c[j]);
  }
#pragma unroll
  for (int j = 0; j < 4; ++j) {
    const float bv = bias[j * 16 + col];
    if (accum) k[j] = k[j] + alpha * (c[j] + bv);
    else       k[j] = c[j] + bv;
  }
}

DEV void mlp(const _Float16* sW0, const _Float16* sW1, const _Float16* sW2,
             const float* sb0, const float* sb1, const float* sb2,
             _Float16* bufA, _Float16* bufB,
             v8f k[4], float alpha, bool accum, int lane) {
  layer_hidden<4, 2, SW0>(bufA, sW0, sb0, bufB, lane);   // [16,64]  -> [16,256]
  layer_hidden<4, 8, SW1>(bufB, sW1, sb1, bufA, lane);   // [16,256] -> [16,256]
  layer_out<SW2>(bufA, sW2, sb2, k, alpha, accum, lane); // [16,256] -> [16,64] regs
}

// Write C-layout f32 regs into the wave's LDS staging buffer as f16 rows.
DEV void store_x(_Float16* buf, const v8f t[4], int lane) {
  const int col = lane & 15;
  const int rb  = (lane & 16) ? 8 : 0;
#pragma unroll
  for (int nt = 0; nt < 4; ++nt)
#pragma unroll
    for (int r = 0; r < 8; ++r)
      buf[(rb + r) * SX + nt * 16 + col] = (_Float16)t[nt][r];
}

__global__ __launch_bounds__(BLOCK, 1)
void ode_tsit5_wmma(const float* __restrict__ ts, const float* __restrict__ y0,
                    const float* __restrict__ gW0, const float* __restrict__ gb0,
                    const float* __restrict__ gW1, const float* __restrict__ gb1,
                    const float* __restrict__ gW2, const float* __restrict__ gb2,
                    float* __restrict__ out) {
  extern __shared__ char smem[];
  _Float16* sW0 = (_Float16*)(smem + OFF_W0);
  _Float16* sW1 = (_Float16*)(smem + OFF_W1);
  _Float16* sW2 = (_Float16*)(smem + OFF_W2);
  float*    sb0 = (float*)(smem + OFF_B0);
  float*    sb1 = (float*)(smem + OFF_B1);
  float*    sb2 = (float*)(smem + OFF_B2);

  const int tid  = threadIdx.x;
  const int lane = tid & 31;
  const int wave = tid >> 5;

  // One-time weight stage: f32 global -> f16 transposed LDS.
  for (int i = tid; i < ND * NW; i += BLOCK) { int k = i / NW, n = i % NW; sW0[n * SW0 + k] = (_Float16)gW0[i]; }
  for (int i = tid; i < NW * NW; i += BLOCK) { int k = i / NW, n = i % NW; sW1[n * SW1 + k] = (_Float16)gW1[i]; }
  for (int i = tid; i < NW * ND; i += BLOCK) { int k = i / ND, n = i % ND; sW2[n * SW2 + k] = (_Float16)gW2[i]; }
  for (int i = tid; i < NW; i += BLOCK) sb0[i] = gb0[i];
  for (int i = tid; i < NW; i += BLOCK) sb1[i] = gb1[i];
  for (int i = tid; i < ND; i += BLOCK) sb2[i] = gb2[i];
  __syncthreads();

  _Float16* bufA = (_Float16*)(smem + OFF_BUF) + (size_t)wave * 2 * (16 * SX);
  _Float16* bufB = bufA + 16 * SX;

  const int col = lane & 15;
  const int rb  = (lane & 16) ? 8 : 0;
  const int rowBase = blockIdx.x * ROWS_PER_BLOCK + wave * 16;

  // y resident in C-layout: y[nt][r] = row (rb+r), col (nt*16+col)
  v8f y[4];
#pragma unroll
  for (int nt = 0; nt < 4; ++nt)
#pragma unroll
    for (int r = 0; r < 8; ++r)
      y[nt][r] = y0[(size_t)(rowBase + rb + r) * ND + nt * 16 + col];

  // save slot t = 0
#pragma unroll
  for (int nt = 0; nt < 4; ++nt)
#pragma unroll
    for (int r = 0; r < 8; ++r)
      out[((size_t)0 * NB + rowBase + rb + r) * ND + nt * 16 + col] = y[nt][r];

#pragma unroll 1
  for (int ti = 0; ti < NT - 1; ++ti) {
    const float dt = ts[ti + 1] - ts[ti];
    const float hh = dt / (float)SUBSTEPS;

#pragma unroll 1
    for (int ss = 0; ss < SUBSTEPS; ++ss) {
      v8f k1[4], k2[4], k3[4], k4[4], k5[4], t[4];

      // stage 1
      store_x(bufA, y, lane);
      mlp(sW0, sW1, sW2, sb0, sb1, sb2, bufA, bufB, k1, 0.0f, false, lane);
      // stage 2
#pragma unroll
      for (int nt = 0; nt < 4; ++nt) t[nt] = y[nt] + (hh * A21) * k1[nt];
      store_x(bufA, t, lane);
      mlp(sW0, sW1, sW2, sb0, sb1, sb2, bufA, bufB, k2, 0.0f, false, lane);
      // stage 3
#pragma unroll
      for (int nt = 0; nt < 4; ++nt) t[nt] = y[nt] + hh * (A31 * k1[nt] + A32 * k2[nt]);
      store_x(bufA, t, lane);
      mlp(sW0, sW1, sW2, sb0, sb1, sb2, bufA, bufB, k3, 0.0f, false, lane);
      // stage 4
#pragma unroll
      for (int nt = 0; nt < 4; ++nt) t[nt] = y[nt] + hh * (A41 * k1[nt] + A42 * k2[nt] + A43 * k3[nt]);
      store_x(bufA, t, lane);
      mlp(sW0, sW1, sW2, sb0, sb1, sb2, bufA, bufB, k4, 0.0f, false, lane);
      // stage 5
#pragma unroll
      for (int nt = 0; nt < 4; ++nt)
        t[nt] = y[nt] + hh * (A51 * k1[nt] + A52 * k2[nt] + A53 * k3[nt] + A54 * k4[nt]);
      store_x(bufA, t, lane);
      mlp(sW0, sW1, sW2, sb0, sb1, sb2, bufA, bufB, k5, 0.0f, false, lane);
      // stage 6 input staged, then fold partial final sum into y,
      // then k6 folded directly during its output layer (never stored).
#pragma unroll
      for (int nt = 0; nt < 4; ++nt)
        t[nt] = y[nt] + hh * (A61 * k1[nt] + A62 * k2[nt] + A63 * k3[nt] + A64 * k4[nt] + A65 * k5[nt]);
      store_x(bufA, t, lane);
#pragma unroll
      for (int nt = 0; nt < 4; ++nt)
        y[nt] = y[nt] + hh * (CB1 * k1[nt] + CB2 * k2[nt] + CB3 * k3[nt] + CB4 * k4[nt] + CB5 * k5[nt]);
      mlp(sW0, sW1, sW2, sb0, sb1, sb2, bufA, bufB, y, hh * CB6, true, lane);
    }

    // save slot ti+1
    float* op = out + ((size_t)(ti + 1) * NB + rowBase) * ND;
#pragma unroll
    for (int nt = 0; nt < 4; ++nt)
#pragma unroll
      for (int r = 0; r < 8; ++r)
        op[(rb + r) * ND + nt * 16 + col] = y[nt][r];
  }
}

extern "C" void kernel_launch(void* const* d_in, const int* in_sizes, int n_in,
                              void* d_out, int out_size, void* d_ws, size_t ws_size,
                              hipStream_t stream) {
  (void)in_sizes; (void)n_in; (void)out_size; (void)d_ws; (void)ws_size;
  const float* ts  = (const float*)d_in[0];
  const float* y0  = (const float*)d_in[1];
  const float* W0  = (const float*)d_in[2];
  const float* b0  = (const float*)d_in[3];
  const float* W1  = (const float*)d_in[4];
  const float* b1  = (const float*)d_in[5];
  const float* W2  = (const float*)d_in[6];
  const float* b2  = (const float*)d_in[7];
  float* out = (float*)d_out;

  ode_tsit5_wmma<<<GRID, BLOCK, LDS_TOTAL, stream>>>(ts, y0, W0, b0, W1, b1, W2, b2, out);
}
